// TransducerJoint_22651657519326
// MI455X (gfx1250) — matrix operand order repaired and checked by reading, not compiled, not run on hardware
//
#include <hip/hip_runtime.h>

typedef __attribute__((ext_vector_type(16))) _Float16 v16h;
typedef __attribute__((ext_vector_type(8)))  _Float16 h8;
typedef __attribute__((ext_vector_type(4)))  _Float16 h4;
typedef __attribute__((ext_vector_type(8)))  float    v8f;

#define B_  4
#define T_  256
#define U_  128
#define DE_ 512
#define DP_ 512
#define D_  640
#define V_  1024

#define BN_ 256                 // N tile per workgroup
#define NT_ 4                   // 16-wide N sub-tiles per wave

// ---------------------------------------------------------------------------
// Branch-free tanh.  gfx1250 has a hardware V_TANH_F32 TRANS op; probe the
// builtin, else use a divergence-free exp2-based identity.
// ---------------------------------------------------------------------------
__device__ __forceinline__ float fast_tanh(float x) {
#if __has_builtin(__builtin_amdgcn_tanhf)
    return __builtin_amdgcn_tanhf(x);
#else
    float ax = __builtin_fabsf(x);
#if __has_builtin(__builtin_amdgcn_exp2f)
    float e = __builtin_amdgcn_exp2f(-2.8853900817779268f * ax);
#else
    float e = exp2f(-2.8853900817779268f * ax);
#endif
#if __has_builtin(__builtin_amdgcn_rcpf)
    float r = 1.0f - 2.0f * e * __builtin_amdgcn_rcpf(1.0f + e);
#else
    float r = 1.0f - 2.0f * e / (1.0f + e);
#endif
    return __builtin_copysignf(r, x);
#endif
}

// ---------------------------------------------------------------------------
// f32 projection: Y[rows,N] = X[rows,K] @ W[K,N] + bias.  One block per row.
// W (1.3MB) stays L2-resident on MI455X (192MB L2).
// ---------------------------------------------------------------------------
__global__ __launch_bounds__(256) void proj_kernel(
    const float* __restrict__ X, const float* __restrict__ W,
    const float* __restrict__ bias, float* __restrict__ Y, int K, int N)
{
    __shared__ float xrow[DE_];
    const int row = blockIdx.x;
    for (int k = threadIdx.x; k < K; k += 256) xrow[k] = X[(long)row * K + k];
    __syncthreads();
    for (int n = threadIdx.x; n < N; n += 256) {
        float acc = bias[n];
        const float* wp = W + n;
#pragma unroll 4
        for (int k = 0; k < K; ++k) acc += xrow[k] * wp[(long)k * N];
        Y[(long)row * N + n] = acc;
    }
}

__global__ __launch_bounds__(256) void cvt_w16(
    const float* __restrict__ W, _Float16* __restrict__ W16, int n)
{
    int i = blockIdx.x * 256 + threadIdx.x;
    if (i < n) W16[i] = (_Float16)W[i];
}

// ---------------------------------------------------------------------------
// Fused lattice GEMM: out[m,n] = sum_k tanh(f[bt(m),k]+g[bu(m),k]) * W16[k,n]
//                               + b_joint[n]
// WG tile 32(M) x 256(N), K step 32, 8 waves; each wave: 1 M sub-tile x 4 N
// sub-tiles -> 4 WMMAs per A fragment.  Fragments are pure 16-byte LDS loads
// matching the ISA 16-bit WMMA VGPR layouts.
// ---------------------------------------------------------------------------
__global__ __launch_bounds__(256) void joint_wmma(
    const float* __restrict__ f, const float* __restrict__ g,
    const _Float16* __restrict__ W16, const float* __restrict__ bj,
    float* __restrict__ out)
{
    __shared__ _Float16 sA [32][40];    // [m][k], 80B rows (16B aligned)
    __shared__ _Float16 sBt[BN_][40];   // [n][k] transposed, 80B rows

    const int tid  = threadIdx.x;
    const int lane = tid & 31;
    const int wave = tid >> 5;
    const int mBase = blockIdx.y * 32;
    const int nBase = blockIdx.x * BN_;

    // ---- A staging coords: thread -> (row fr, k-quad fk) of the 32x32 tile
    const int fr = tid >> 3;          // 0..31
    const int fk = (tid & 7) * 4;     // 0,4,..,28
    const int m0  = mBase + fr;
    const int bb  = m0 >> 15;          // / (T_*U_)
    const int rem = m0 & 32767;
    const int tt  = rem >> 7;          // / U_
    const int uu  = rem & 127;         // % U_
    const float* fptr = f + (size_t)(bb * T_ + tt) * D_ + fk;
    const float* gptr = g + (size_t)(bb * U_ + uu) * D_ + fk;

    // ---- wave tile assignment: 2 (M) x 4 (N-quad) waves
    const int waveRow = (wave & 1) * 16;        // M sub-tile
    const int waveCol = (wave >> 1) * (NT_ * 16); // base of 4 N sub-tiles
    const int rowA = lane & 15;
    const int hi   = lane >> 4;
    const int col  = lane & 15;
    const int koff = hi * 8;

    v8f acc[NT_] = {};

    for (int kt = 0; kt < D_; kt += 32) {
        __syncthreads();

        // ---- stage B tile transposed: coalesced b128 global, b16 LDS scatter
#pragma unroll
        for (int j = 0; j < 4; ++j) {
            int idx = j * 256 + tid;   // 0..1023 over 32K x 256N in 8-half chunks
            int kr  = idx >> 5;        // K row 0..31
            int ch  = idx & 31;        // 8-half chunk along N
            const _Float16* sp = W16 + (size_t)(kt + kr) * V_ + nBase + ch * 8;
            h8 w = *(const h8*)sp;
#pragma unroll
            for (int i = 0; i < 8; ++i) sBt[ch * 8 + i][kr] = w[i];
        }

        // ---- stage A tile: float4 f + float4 g -> 4x tanh -> b64 LDS store
        {
            float4 fv = *(const float4*)(fptr + kt);
            float4 gv = *(const float4*)(gptr + kt);
            h4 av;
            av[0] = (_Float16)fast_tanh(fv.x + gv.x);
            av[1] = (_Float16)fast_tanh(fv.y + gv.y);
            av[2] = (_Float16)fast_tanh(fv.z + gv.z);
            av[3] = (_Float16)fast_tanh(fv.w + gv.w);
            *(h4*)&sA[fr][fk] = av;
        }
        __syncthreads();

        // ---- A fragment: two contiguous 8-half runs (ISA 16-bit A layout)
        const _Float16* arow = &sA[waveRow + rowA][0];
        h8 a_lo = *(const h8*)(arow + hi * 8);
        h8 a_hi = *(const h8*)(arow + 16 + hi * 8);
        v16h a = __builtin_shufflevector(a_lo, a_hi,
                     0,1,2,3,4,5,6,7,8,9,10,11,12,13,14,15);

        // ---- B fragments: two 8-half K runs (k = 8*hi.., 16+8*hi..), matching A
#pragma unroll
        for (int tN = 0; tN < NT_; ++tN) {
            const _Float16* bp = &sBt[waveCol + tN * 16 + col][koff];
            h8 b_lo = *(const h8*)(bp);
            h8 b_hi = *(const h8*)(bp + 16);
            v16h b = __builtin_shufflevector(b_lo, b_hi,
                         0,1,2,3,4,5,6,7,8,9,10,11,12,13,14,15);
            acc[tN] = __builtin_amdgcn_wmma_f32_16x16x32_f16(
                          false, a, false, b, (short)0, acc[tN], false, false);
        }
    }

    // ---- epilogue: C/D layout (VGPR i: M = i + 8*hi, N = col) + bias.
    // Output is 537MB write-once: NT stores keep W16/f/g resident in L2.
#pragma unroll
    for (int tN = 0; tN < NT_; ++tN) {
        const int n0 = nBase + waveCol + tN * 16 + col;
        const float bjv = bj[n0];
#pragma unroll
        for (int i = 0; i < 8; ++i) {
            int m = mBase + waveRow + hi * 8 + i;
            __builtin_nontemporal_store(acc[tN][i] + bjv,
                                        &out[(size_t)m * V_ + n0]);
        }
    }
}

// ---------------------------------------------------------------------------
extern "C" void kernel_launch(void* const* d_in, const int* in_sizes, int n_in,
                              void* d_out, int out_size, void* d_ws, size_t ws_size,
                              hipStream_t stream)
{
    const float* enc     = (const float*)d_in[0];
    const float* pred    = (const float*)d_in[1];
    const float* W_enc   = (const float*)d_in[2];
    const float* b_enc   = (const float*)d_in[3];
    const float* W_pred  = (const float*)d_in[4];
    const float* b_pred  = (const float*)d_in[5];
    const float* W_joint = (const float*)d_in[6];
    const float* b_joint = (const float*)d_in[7];
    float* out = (float*)d_out;

    // workspace: f (B*T*D f32) | g (B*U*D f32) | W16 (D*V f16)  ~5.2 MB
    char* ws = (char*)d_ws;
    float*    fbuf = (float*)ws;
    float*    gbuf = (float*)(ws + sizeof(float) * (size_t)(B_ * T_ * D_));
    _Float16* w16  = (_Float16*)(ws + sizeof(float) * (size_t)(B_ * T_ * D_ + B_ * U_ * D_));

    proj_kernel<<<dim3(B_ * T_), dim3(256), 0, stream>>>(enc,  W_enc,  b_enc,  fbuf, DE_, D_);
    proj_kernel<<<dim3(B_ * U_), dim3(256), 0, stream>>>(pred, W_pred, b_pred, gbuf, DP_, D_);
    cvt_w16<<<dim3((D_ * V_ + 255) / 256), dim3(256), 0, stream>>>(W_joint, w16, D_ * V_);

    joint_wmma<<<dim3(V_ / BN_, (B_ * T_ * U_) / 32), dim3(256), 0, stream>>>(
        fbuf, gbuf, w16, b_joint, out);
}
